// SelfAttentive_10393820856442
// MI455X (gfx1250) — matrix operand order repaired and checked by reading, not compiled, not run on hardware
//
#include <hip/hip_runtime.h>
#include <hip/hip_bf16.h>
#include <math.h>

// ---------------- problem constants ----------------
// NTOKEN=50000 NINP=300 NHID=512 DA=350 R=30 MLP_NHID=2000 NCLASS=5 B=64 T=512
#define CB    64
#define CT    512
#define CMT   32768     // B*T
#define CNIN  300
#define CKIN  320       // NINP padded to mult of 32
#define CH    512
#define CG    2048      // 4*NHID
#define CH2   1024      // 2*NHID
#define CDAP  352       // DA=350 padded to mult of 32
#define CDAPR 384       // DA padded to mult of 128 (rows of W1 tile space)
#define CRR   30
#define CRP   32
#define CRPR  128       // W2 rows padded to tile
#define CMH   2000
#define CMHR  2048      // Wm rows padded to tile
#define CKBM  30720     // R*2H
#define CNC   5

// CDNA5 async global->LDS path (ASYNCcnt), guarded so absence = safe fallback.
#if defined(__AMDGCN__) && __has_builtin(__builtin_amdgcn_global_load_async_to_lds_b128) && \
    __has_builtin(__builtin_amdgcn_s_wait_asynccnt)
#define USE_ASYNC_LDS 1
#else
#define USE_ASYNC_LDS 0
#endif
#define GAS __attribute__((address_space(1)))
#define LAS __attribute__((address_space(3)))
typedef int async_i4 __attribute__((__vector_size__(4 * sizeof(int))));

// 16-byte async global->LDS transfer (GLOBAL_LOAD_ASYNC_TO_LDS_B128).
__device__ __forceinline__ void async_ld_b128(void* lds_dst, const void* gsrc) {
#if USE_ASYNC_LDS
  __builtin_amdgcn_global_load_async_to_lds_b128((GAS async_i4*)gsrc,
                                                 (LAS async_i4*)lds_dst, 0, 0);
#else
  (void)lds_dst; (void)gsrc;
#endif
}
__device__ __forceinline__ void async_wait0() {
#if USE_ASYNC_LDS
  __builtin_amdgcn_s_wait_asynccnt(0);
#endif
}

typedef __attribute__((ext_vector_type(16))) __bf16 v16bf;
typedef __attribute__((ext_vector_type(8)))  float  v8f;

// ---------------- WMMA fragment helpers (A/B in contiguous k = 16*half + s order; C/D per CDNA5 ISA) ---
__device__ __forceinline__ v8f wmma_bf16(v16bf a, v16bf b, v8f c) {
  return __builtin_amdgcn_wmma_f32_16x16x32_bf16(false, a, false, b, (short)0, c,
                                                 false, false);
}

// A 16x32 bf16 from row-major LDS tile t[row][k], row stride ld (elements).
__device__ __forceinline__ v16bf load_frag_a(const __bf16* t, int row0, int k0, int ld) {
  const int lane = threadIdx.x & 31;
  const int m = lane & 15, half = lane >> 4;
  const __bf16* base = t + (row0 + m) * ld + k0 + 16 * half;
  v16bf a;
#pragma unroll
  for (int v = 0; v < 8; ++v) {
    const int kk = 2 * v;
    union { unsigned u; __bf16 h[2]; } cv;
    cv.u = *(const unsigned*)(base + kk);
    a[2 * v] = cv.h[0];
    a[2 * v + 1] = cv.h[1];
  }
  return a;
}

// B 32x16 bf16, logical B[k][n] = W[n][k], source tile stored row-major [n][k].
__device__ __forceinline__ v16bf load_frag_b(const __bf16* t, int n0, int k0, int ld) {
  const int lane = threadIdx.x & 31;
  const int n = lane & 15, half = lane >> 4;
  const __bf16* base = t + (n0 + n) * ld + k0 + 16 * half;
  v16bf b;
#pragma unroll
  for (int v = 0; v < 8; ++v) {
    union { unsigned u; __bf16 h[2]; } cv;
    cv.u = *(const unsigned*)(base + 2 * v);
    b[2 * v] = cv.h[0];
    b[2 * v + 1] = cv.h[1];
  }
  return b;
}

// B 32x16 bf16 from tile stored row-major [k][n] (true B layout).
__device__ __forceinline__ v16bf load_frag_b_kn(const __bf16* t, int k0, int n0, int ld) {
  const int lane = threadIdx.x & 31;
  const int n = lane & 15, half = lane >> 4;
  const __bf16* base = t + (k0 + 16 * half) * ld + n0 + n;
  v16bf b;
#pragma unroll
  for (int v = 0; v < 8; ++v) {
    b[2 * v] = base[(2 * v) * ld];
    b[2 * v + 1] = base[(2 * v + 1) * ld];
  }
  return b;
}

// D/C 16x16 f32 store, row stride ld.
__device__ __forceinline__ void store_cfrag(float* dst, v8f c, int row0, int col0, int ld) {
  const int lane = threadIdx.x & 31;
  const int n = lane & 15, half = lane >> 4;
#pragma unroll
  for (int e = 0; e < 8; ++e) dst[(row0 + half * 8 + e) * ld + col0 + n] = c[e];
}

// ---------------- kernel: f32 -> bf16 convert with zero padding --------------
__global__ void cvt_pad_kernel(const float* __restrict__ src, __bf16* __restrict__ dst,
                               int srcR, int srcC, int dstC, long total) {
  for (long i = (long)blockIdx.x * blockDim.x + threadIdx.x; i < total;
       i += (long)gridDim.x * blockDim.x) {
    const int r = (int)(i / dstC), c = (int)(i % dstC);
    float v = (r < srcR && c < srcC) ? src[(size_t)r * srcC + c] : 0.f;
    dst[i] = (__bf16)v;
  }
}

// ---------------- kernel: embedding gather + mask + in-length reversal -------
__global__ void embed_kernel(const int* __restrict__ ids, const int* __restrict__ lens,
                             const float* __restrict__ emb,
                             __bf16* __restrict__ xf, __bf16* __restrict__ xb) {
  const int row = blockIdx.x;            // b*T + t
  const int b = row >> 9, t = row & 511;
  const int len = lens[b];
  const bool valid = t < len;
  const int ts = valid ? (len - 1 - t) : t;
  const float* ef = emb + (size_t)ids[row] * CNIN;
  const float* eb = emb + (size_t)ids[(b << 9) + ts] * CNIN;
  __bf16* of = xf + (size_t)row * CKIN;
  __bf16* ob = xb + (size_t)row * CKIN;
  for (int d = threadIdx.x; d < CKIN; d += blockDim.x) {
    const bool in = valid && (d < CNIN);
    of[d] = (__bf16)(in ? ef[d] : 0.f);
    ob[d] = (__bf16)(in ? eb[d] : 0.f);
  }
}

// ---------------- generic NT GEMM: C[M,N] = A[M,K] @ W[N,K]^T ----------------
// mode 0: f32 store (+bias)   mode 1: bf16 store of tanh(v)   mode 2: f32 relu(v+bias)
// Async path requires: W rows padded (zeros) to 128-multiples; A-row overreads
// must stay inside the workspace (guaranteed by padded allocations).
#define GT_LD 40  // 32 + 8 pad (bf16 elements)
__global__ __launch_bounds__(256) void gemm_nt_kernel(
    const __bf16* __restrict__ A, const __bf16* __restrict__ W,
    float* __restrict__ outF, __bf16* __restrict__ outBf, const float* __restrict__ bias,
    int M, int N, int K, int ldc, int mode) {
  __shared__ __bf16 As[128 * GT_LD];
  __shared__ __bf16 Ws[128 * GT_LD];
  const int m0 = blockIdx.y * 128;
  const int n0 = blockIdx.x * 128;
  const int tid = threadIdx.x;
  const int w = tid >> 5;
  const int mw = (w & 1) * 4;   // wave's first m-subtile
  const int nw = (w >> 1) * 2;  // wave's first n-subtile
  v8f acc[4][2] = {};
  const int sr = tid >> 1, scb = (tid & 1) * 16;  // staging: row, 16-elem half
  for (int k0 = 0; k0 < K; k0 += 32) {
#if USE_ASYNC_LDS
    {
      const __bf16* gA = A + (size_t)(m0 + sr) * K + k0 + scb;
      const __bf16* gW = W + (size_t)(n0 + sr) * K + k0 + scb;
      __bf16* lA = &As[sr * GT_LD + scb];
      __bf16* lW = &Ws[sr * GT_LD + scb];
      async_ld_b128(lA, gA);
      async_ld_b128(lA + 8, gA + 8);
      async_ld_b128(lW, gW);
      async_ld_b128(lW + 8, gW + 8);
      async_wait0();
    }
    __syncthreads();
#else
    {
      uint4 z = {0u, 0u, 0u, 0u};
      uint4 a0 = z, a1 = z, b0 = z, b1 = z;
      if (m0 + sr < M) {
        const uint4* p = (const uint4*)(A + (size_t)(m0 + sr) * K + k0 + scb);
        a0 = p[0]; a1 = p[1];
        if (k0 + 32 < K) __builtin_prefetch(A + (size_t)(m0 + sr) * K + k0 + 32 + scb, 0, 1);
      }
      if (n0 + sr < N) {
        const uint4* p = (const uint4*)(W + (size_t)(n0 + sr) * K + k0 + scb);
        b0 = p[0]; b1 = p[1];
        if (k0 + 32 < K) __builtin_prefetch(W + (size_t)(n0 + sr) * K + k0 + 32 + scb, 0, 1);
      }
      *(uint4*)(&As[sr * GT_LD + scb]) = a0;
      *(uint4*)(&As[sr * GT_LD + scb + 8]) = a1;
      *(uint4*)(&Ws[sr * GT_LD + scb]) = b0;
      *(uint4*)(&Ws[sr * GT_LD + scb + 8]) = b1;
    }
    __syncthreads();
#endif
    v16bf af[4], bf[2];
#pragma unroll
    for (int i = 0; i < 4; ++i) af[i] = load_frag_a(As, (mw + i) * 16, 0, GT_LD);
#pragma unroll
    for (int j = 0; j < 2; ++j) bf[j] = load_frag_b(Ws, (nw + j) * 16, 0, GT_LD);
#pragma unroll
    for (int i = 0; i < 4; ++i)
#pragma unroll
      for (int j = 0; j < 2; ++j) acc[i][j] = wmma_bf16(af[i], bf[j], acc[i][j]);
    __syncthreads();
  }
  const int lane = tid & 31, nn = lane & 15, half = lane >> 4;
#pragma unroll
  for (int i = 0; i < 4; ++i)
#pragma unroll
    for (int j = 0; j < 2; ++j) {
      const int c = n0 + (nw + j) * 16 + nn;
      if (c >= N) continue;
      const float bv = bias ? bias[c] : 0.f;
      const int rbase = m0 + (mw + i) * 16 + half * 8;
#pragma unroll
      for (int e = 0; e < 8; ++e) {
        const int r = rbase + e;
        if (r >= M) break;
        const float v = acc[i][j][e] + bv;
        if (mode == 1)      outBf[(size_t)r * ldc + c] = (__bf16)tanhf(v);
        else if (mode == 2) outF[(size_t)r * ldc + c] = v > 0.f ? v : 0.f;
        else                outF[(size_t)r * ldc + c] = v;
      }
    }
}

// ---------------- LSTM recurrence (both directions) --------------------------
// 32 workgroups: dir = blk>>4, slice = blk&15 (32 hidden units / slice).
// W_hh slice (128 gate-rows x 512, bf16) stays in LDS for all 512 steps.
// Per step: G = h_prev @ Wslice^T via WMMA, gates, ping-pong h, global barrier.
#define LW 520  // 512 + 8 pad
__global__ __launch_bounds__(256) void lstm_kernel(
    const float* __restrict__ WhhF, const float* __restrict__ WhhB,
    const float* __restrict__ xgF, const float* __restrict__ xgB,
    const int* __restrict__ lens, __bf16* __restrict__ hbuf,
    unsigned* __restrict__ cnt, __bf16* __restrict__ Hout) {
  extern __shared__ char smem[];
  __bf16* Wsl = (__bf16*)smem;            // [128][LW]
  __bf16* hs  = Wsl + 128 * LW;           // [64][LW]
  float*  Gs  = (float*)(hs + 64 * LW);   // [64][136]
  float*  cs  = Gs + 64 * 136;            // [64][32]
  const int dir = blockIdx.x >> 4;
  const int slice = blockIdx.x & 15;
  const int h0 = slice * 32;
  const float* Whh = dir ? WhhB : WhhF;
  const float* XG  = dir ? xgB : xgF;
  __bf16* hb0 = hbuf + (size_t)dir * 2 * CB * CH;
  __bf16* hb1 = hb0 + CB * CH;
  unsigned* ctr = cnt + dir;
  const int tid = threadIdx.x;
  const int w = tid >> 5;

  // Preload + convert W_hh slice into LDS (weight-stationary across 512 steps).
  for (int e = tid; e < (128 * 512) / 4; e += 256) {
    const int rr = e >> 7, k4 = (e & 127) * 4;
    const int gate = rr >> 5, u = rr & 31;
    const float4 v = *(const float4*)(Whh + (size_t)(gate * 512 + h0 + u) * CH + k4);
    union { __bf16 h[4]; uint2 u2; } cv;
    cv.h[0] = (__bf16)v.x; cv.h[1] = (__bf16)v.y;
    cv.h[2] = (__bf16)v.z; cv.h[3] = (__bf16)v.w;
    *(uint2*)(&Wsl[rr * LW + k4]) = cv.u2;
  }
  for (int e = tid; e < 64 * 32; e += 256) cs[e] = 0.f;
  __builtin_amdgcn_s_wait_tensorcnt(0);
  __syncthreads();

  for (int t = 0; t < CT; ++t) {
    __bf16* hprev = (t & 1) ? hb1 : hb0;
    __bf16* hnext = (t & 1) ? hb0 : hb1;
    for (int e = tid; e < (CB * CH) / 8; e += 256) {
      const int bb = e >> 6, k8 = (e & 63) * 8;
#if USE_ASYNC_LDS
      async_ld_b128(&hs[bb * LW + k8], hprev + (size_t)bb * CH + k8);
#else
      *(uint4*)(&hs[bb * LW + k8]) = *(const uint4*)(hprev + (size_t)bb * CH + k8);
#endif
    }
    async_wait0();
    __syncthreads();
    // G[64 x 128] = h_prev[64 x 512] @ Wslice[128 x 512]^T ; wave w owns n-tile w.
    v8f acc[4] = {};
    for (int kc = 0; kc < 512; kc += 32) {
      const v16bf bf = load_frag_b(Wsl, w * 16, kc, LW);
#pragma unroll
      for (int mt = 0; mt < 4; ++mt) {
        const v16bf af = load_frag_a(hs, mt * 16, kc, LW);
        acc[mt] = wmma_bf16(af, bf, acc[mt]);
      }
    }
#pragma unroll
    for (int mt = 0; mt < 4; ++mt) store_cfrag(Gs, acc[mt], mt * 16, w * 16, 136);
    __syncthreads();
    // gates: i,f,g,o (PyTorch order, 512 cols apart in full 2048 layout)
    for (int p = tid; p < CB * 32; p += 256) {
      const int bb = p >> 5, u = p & 31;
      const size_t xr = ((size_t)bb * CT + t) * CG + h0 + u;
      const float gi = Gs[bb * 136 + u]      + XG[xr];
      const float gf = Gs[bb * 136 + 32 + u] + XG[xr + 512];
      const float gg = Gs[bb * 136 + 64 + u] + XG[xr + 1024];
      const float go = Gs[bb * 136 + 96 + u] + XG[xr + 1536];
      const float si = 1.f / (1.f + expf(-gi));
      const float sf = 1.f / (1.f + expf(-gf));
      const float so = 1.f / (1.f + expf(-go));
      const float cn = sf * cs[p] + si * tanhf(gg);
      const float hh = so * tanhf(cn);
      cs[p] = cn;
      hnext[bb * CH + h0 + u] = (__bf16)hh;
      const int len = lens[bb];
      const int pos = dir ? ((t < len) ? (len - 1 - t) : t) : t;
      Hout[((size_t)bb * CT + pos) * CH2 + dir * CH + h0 + u] = (__bf16)hh;
    }
    // global barrier across this direction's 16 workgroups
    __syncthreads();
    if (tid == 0) {
      __threadfence();
      atomicAdd(ctr, 1u);
      const unsigned target = 16u * (unsigned)(t + 1);
      while (atomicAdd(ctr, 0u) < target) __builtin_amdgcn_s_sleep(1);
      __threadfence();
    }
    __syncthreads();
  }
}

// ---------------- masked softmax over T for each (b, r) ----------------------
__global__ __launch_bounds__(256) void softmax_kernel(
    const float* __restrict__ s2, const int* __restrict__ lens,
    float* __restrict__ A, __bf16* __restrict__ Abf) {
  __shared__ float red[256];
  const int blk = blockIdx.x;
  const int b = blk / CRR, r = blk % CRR;
  const int len = lens[b];
  const int tid = threadIdx.x;
  const int t0 = tid, t1 = tid + 256;
  float v0 = -1e30f, v1 = -1e30f;
  if (t0 < len) v0 = s2[((size_t)(b * CT + t0)) * CRP + r];
  if (t1 < len) v1 = s2[((size_t)(b * CT + t1)) * CRP + r];
  red[tid] = fmaxf(v0, v1);
  __syncthreads();
  for (int s = 128; s > 0; s >>= 1) {
    if (tid < s) red[tid] = fmaxf(red[tid], red[tid + s]);
    __syncthreads();
  }
  const float m = red[0];
  __syncthreads();
  const float e0 = (t0 < len) ? expf(v0 - m) : 0.f;
  const float e1 = (t1 < len) ? expf(v1 - m) : 0.f;
  red[tid] = e0 + e1;
  __syncthreads();
  for (int s = 128; s > 0; s >>= 1) {
    if (tid < s) red[tid] += red[tid + s];
    __syncthreads();
  }
  const float inv = 1.f / red[0];
  const size_t o = ((size_t)b * CRR + r) * CT;
  const size_t ob = ((size_t)b * CRP + r) * CT;
  A[o + t0] = e0 * inv;
  A[o + t1] = e1 * inv;
  Abf[ob + t0] = (__bf16)(e0 * inv);
  Abf[ob + t1] = (__bf16)(e1 * inv);
}

// ---------------- M = A @ Hout per batch, packed into BM[b][r*1024+n] --------
__global__ __launch_bounds__(256) void attn_m_kernel(
    const __bf16* __restrict__ Abf, const __bf16* __restrict__ Hout,
    __bf16* __restrict__ BMo) {
  __shared__ __bf16 As[32 * GT_LD];
  __shared__ __bf16 Bs[32 * 136];
  const int b = blockIdx.y, n0g = blockIdx.x * 128;
  const int tid = threadIdx.x, w = tid >> 5;
  const __bf16* Ab = Abf + (size_t)b * CRP * CT;
  const __bf16* Hb = Hout + (size_t)b * CT * CH2;
  v8f acc[2] = {};
  const int ar = tid >> 3, ac = (tid & 7) * 4;    // A staging: 32x32
  const int br = tid >> 3, bc = (tid & 7) * 16;   // B staging: 32x128
  for (int k0 = 0; k0 < CT; k0 += 32) {
    *(uint2*)(&As[ar * GT_LD + ac]) = *(const uint2*)(Ab + (size_t)ar * CT + k0 + ac);
    const __bf16* hsrc = Hb + (size_t)(k0 + br) * CH2 + n0g + bc;
#if USE_ASYNC_LDS
    async_ld_b128(&Bs[br * 136 + bc], hsrc);
    async_ld_b128(&Bs[br * 136 + bc + 8], hsrc + 8);
    async_wait0();
#else
    *(uint4*)(&Bs[br * 136 + bc]) = *(const uint4*)(hsrc);
    *(uint4*)(&Bs[br * 136 + bc + 8]) = *(const uint4*)(hsrc + 8);
#endif
    __syncthreads();
    const v16bf bf = load_frag_b_kn(Bs, 0, w * 16, 136);
#pragma unroll
    for (int mt = 0; mt < 2; ++mt) {
      const v16bf af = load_frag_a(As, mt * 16, 0, GT_LD);
      acc[mt] = wmma_bf16(af, bf, acc[mt]);
    }
    __syncthreads();
  }
  const int lane = tid & 31, nn = lane & 15, half = lane >> 4;
#pragma unroll
  for (int mt = 0; mt < 2; ++mt)
#pragma unroll
    for (int e = 0; e < 8; ++e) {
      const int r = mt * 16 + half * 8 + e;
      if (r < CRR)
        BMo[(size_t)b * CKBM + (size_t)r * CH2 + n0g + w * 16 + nn] = (__bf16)acc[mt][e];
    }
}

// ---------------- penalty: ||A A^T - I||_F^2 per batch -----------------------
__global__ __launch_bounds__(256) void penalty_kernel(const float* __restrict__ A,
                                                      float* __restrict__ part) {
  __shared__ float red[256];
  const int b = blockIdx.x, tid = threadIdx.x;
  const float* Ab = A + (size_t)b * CRR * CT;
  float local = 0.f;
  for (int p = tid; p < CRR * CRR; p += 256) {
    const int r = p / CRR, s = p % CRR;
    const float* ar = Ab + (size_t)r * CT;
    const float* as = Ab + (size_t)s * CT;
    float d = 0.f;
    for (int t = 0; t < CT; ++t) d += ar[t] * as[t];
    d -= (r == s) ? 1.f : 0.f;
    local += d * d;
  }
  red[tid] = local;
  __syncthreads();
  for (int s = 128; s > 0; s >>= 1) {
    if (tid < s) red[tid] += red[tid + s];
    __syncthreads();
  }
  if (tid == 0) part[b] = red[0];
}

// ---------------- final: decoded = act @ Wd^T + bd ; penal -------------------
__global__ void finish_kernel(const float* __restrict__ act, const float* __restrict__ Wd,
                              const float* __restrict__ bd, const float* __restrict__ part,
                              float* __restrict__ out) {
  const int tid = threadIdx.x;
  if (tid < CB * CNC) {
    const int b = tid / CNC, j = tid % CNC;
    const float* a = act + (size_t)b * CMH;
    const float* wv = Wd + (size_t)j * CMH;
    float s = bd[j];
    for (int k = 0; k < CMH; ++k) s += a[k] * wv[k];
    out[tid] = s;
  }
  if (tid == CB * CNC) {
    float s = 0.f;
    for (int i = 0; i < CB; ++i) s += part[i];
    out[CB * CNC] = s / (float)CB;
  }
}

// ---------------- host launcher ----------------------------------------------
extern "C" void kernel_launch(void* const* d_in, const int* in_sizes, int n_in,
                              void* d_out, int out_size, void* d_ws, size_t ws_size,
                              hipStream_t stream) {
  (void)in_sizes; (void)n_in; (void)out_size; (void)ws_size;
  const int*   input_ids = (const int*)d_in[0];
  const int*   len_li    = (const int*)d_in[1];
  const float* emb       = (const float*)d_in[2];
  const float* W_ih_f    = (const float*)d_in[3];
  const float* W_hh_f    = (const float*)d_in[4];
  const float* W_ih_b    = (const float*)d_in[5];
  const float* W_hh_b    = (const float*)d_in[6];
  const float* W1        = (const float*)d_in[7];
  const float* W2        = (const float*)d_in[8];
  const float* Wm        = (const float*)d_in[9];
  const float* bm        = (const float*)d_in[10];
  const float* Wd        = (const float*)d_in[11];
  const float* bd        = (const float*)d_in[12];
  float* out = (float*)d_out;

  // workspace carve-up (256B aligned); all GEMM operand buffers padded to full
  // 128-row tiles so the async-LDS staging path never reads out of bounds.
  char* base = (char*)d_ws;
  size_t off = 0;
  auto alloc = [&](size_t bytes) -> char* {
    char* p = base + off;
    off = (off + bytes + 255) & ~(size_t)255;
    return p;
  };
  __bf16* xf      = (__bf16*)alloc((size_t)CMT * CKIN * 2);
  __bf16* xb      = (__bf16*)alloc((size_t)CMT * CKIN * 2);
  __bf16* wihf    = (__bf16*)alloc((size_t)CG * CKIN * 2);
  __bf16* wihb    = (__bf16*)alloc((size_t)CG * CKIN * 2);
  __bf16* w1b     = (__bf16*)alloc((size_t)CDAPR * CH2 * 2);
  __bf16* w2b     = (__bf16*)alloc((size_t)CRPR * CDAP * 2);
  __bf16* wmb     = (__bf16*)alloc((size_t)CMHR * CKBM * 2);
  float*  xgf     = (float*)alloc((size_t)CMT * CG * 4);
  float*  xgb     = (float*)alloc((size_t)CMT * CG * 4);
  __bf16* hbuf    = (__bf16*)alloc((size_t)2 * 2 * CB * CH * 2);  // zeroed
  unsigned* cnt   = (unsigned*)alloc(256);                        // zeroed
  __bf16* houtb   = (__bf16*)alloc((size_t)CMT * CH2 * 2);
  __bf16* tmpa    = (__bf16*)alloc((size_t)CMT * CDAP * 2);
  float*  s2      = (float*)alloc((size_t)CMT * CRP * 4);
  float*  Amat    = (float*)alloc((size_t)CB * CRR * CT * 4);
  __bf16* Abf     = (__bf16*)alloc((size_t)CB * CRP * CT * 2);    // zeroed (pad rows)
  __bf16* BMo     = (__bf16*)alloc((size_t)128 * CKBM * 2);       // 64 rows + tile pad
  float*  act     = (float*)alloc((size_t)CB * CMH * 4);
  float*  part    = (float*)alloc((size_t)CB * 4);

  // zero-init state buffers (deterministic per call)
  (void)hipMemsetAsync(hbuf, 0, (size_t)2 * 2 * CB * CH * 2, stream);
  (void)hipMemsetAsync(cnt, 0, 256, stream);
  (void)hipMemsetAsync(Abf, 0, (size_t)CB * CRP * CT * 2, stream);

  // weight conversions to bf16 (zero-padded to full tiles)
  cvt_pad_kernel<<<1024, 256, 0, stream>>>(W_ih_f, wihf, CG, CNIN, CKIN, (long)CG * CKIN);
  cvt_pad_kernel<<<1024, 256, 0, stream>>>(W_ih_b, wihb, CG, CNIN, CKIN, (long)CG * CKIN);
  cvt_pad_kernel<<<1024, 256, 0, stream>>>(W1, w1b, 350, CH2, CH2, (long)CDAPR * CH2);
  cvt_pad_kernel<<<64, 256, 0, stream>>>(W2, w2b, CRR, 350, CDAP, (long)CRPR * CDAP);
  cvt_pad_kernel<<<4096, 256, 0, stream>>>(Wm, wmb, CMH, CKBM, CKBM, (long)CMHR * CKBM);

  // embedding gather (+mask, +within-length reversal for backward direction)
  embed_kernel<<<CMT, 256, 0, stream>>>(input_ids, len_li, emb, xf, xb);

  // input-gate GEMMs: XG = x @ W_ih^T  [32768 x 2048], K=320
  gemm_nt_kernel<<<dim3(CG / 128, CMT / 128), 256, 0, stream>>>(
      xf, wihf, xgf, nullptr, nullptr, CMT, CG, CKIN, CG, 0);
  gemm_nt_kernel<<<dim3(CG / 128, CMT / 128), 256, 0, stream>>>(
      xb, wihb, xgb, nullptr, nullptr, CMT, CG, CKIN, CG, 0);

  // bidirectional LSTM recurrence (32 cooperating WGs, LDS-resident weights)
  const int lstm_smem = (128 * LW + 64 * LW) * 2 + (64 * 136 + 64 * 32) * 4;
  (void)hipFuncSetAttribute((const void*)lstm_kernel,
                            hipFuncAttributeMaxDynamicSharedMemorySize, lstm_smem);
  lstm_kernel<<<32, 256, lstm_smem, stream>>>(W_hh_f, W_hh_b, xgf, xgb, len_li, hbuf,
                                              cnt, houtb);

  // attention scores: tmp = tanh(Hout @ W1^T) [32768 x 352], then s2 = tmp @ W2^T
  gemm_nt_kernel<<<dim3(CDAPR / 128, CMT / 128), 256, 0, stream>>>(
      houtb, w1b, nullptr, tmpa, nullptr, CMT, CDAP, CH2, CDAP, 1);
  gemm_nt_kernel<<<dim3(1, CMT / 128), 256, 0, stream>>>(
      tmpa, w2b, s2, nullptr, nullptr, CMT, CRP, CDAP, CRP, 0);

  // masked softmax over time
  softmax_kernel<<<CB * CRR, 256, 0, stream>>>(s2, len_li, Amat, Abf);

  // M = A @ Hout per batch -> BM (bf16, packed [B, 30720])
  attn_m_kernel<<<dim3(CH2 / 128, CB), 256, 0, stream>>>(Abf, houtb, BMo);

  // penalization term
  penalty_kernel<<<CB, 256, 0, stream>>>(Amat, part);

  // MLP layer 1: act = relu(BM @ Wm^T + bm)  [64 x 2000], K=30720
  gemm_nt_kernel<<<dim3(CMHR / 128, 1), 256, 0, stream>>>(
      BMo, wmb, act, nullptr, bm, CB, CMH, CKBM, CMH, 2);

  // MLP layer 2 + penal reduce -> d_out[0:320], d_out[320]
  finish_kernel<<<1, 512, 0, stream>>>(act, Wd, bd, part, out);
}
